// SeperateLstmSumAlignableAgent_29051158790178
// MI455X (gfx1250) — matrix operand order repaired and checked by reading, not compiled, not run on hardware
//
#include <hip/hip_runtime.h>
#include <hip/hip_bf16.h>
#include <cstdint>
#include <cstddef>

// ---------------------------------------------------------------------------
// Types for CDNA5 WMMA (wave32): D(f32 16x16) = A(bf16 16x32) x B(bf16 32x16) + C
// ---------------------------------------------------------------------------
typedef __attribute__((ext_vector_type(16))) __bf16 v16bf;
typedef __attribute__((ext_vector_type(8)))  float  v8f;

static constexpr int T_   = 128;
static constexpr int B_   = 32;
static constexpr int NF_  = 4096;   // T*B frames
static constexpr int CH_  = 256;    // frames per pipeline chunk
static constexpr int NCHUNK_ = NF_ / CH_;

#define DIVUP(a, b) (((a) + (b) - 1) / (b))

// A-fragment k map (16-bit A 16x32, ISA 7.12.2): element e of lane's v16bf
__device__ __forceinline__ int kmapA(int lane, int e) {
    return (e & 7) + ((e >> 3) << 4) + ((lane >> 4) << 3);
}
// B-fragment k map (16-bit B 32x16): same k order as A (e<8: K=8h+e, e>=8: K=16+8h+e-8)
__device__ __forceinline__ int kmapB(int lane, int e) {
    return (e & 7) + ((e >> 3) << 4) + ((lane >> 4) << 3);
}

__device__ __forceinline__ v8f wmma_bf16(v16bf a, v16bf b, v8f c) {
    return __builtin_amdgcn_wmma_f32_16x16x32_bf16(false, a, false, b, (short)0, c,
                                                   false, false);
}

__device__ __forceinline__ float sigm(float x) { return 1.0f / (1.0f + __expf(-x)); }

// ---------------------------------------------------------------------------
// Weight prepack: src f32 [Nout][Cin][KH][KW] (plain [Nout][K] when KH=KW=1)
// -> packed B fragments: [(tileN*Ksteps+ks)*32 + lane]*16 + e
// im2col k-order: k = (ky*KW + kx)*Cin + ci
// ---------------------------------------------------------------------------
__global__ void pack_b_kernel(const float* __restrict__ src, __bf16* __restrict__ dst,
                              int Nout, int Cin, int KH, int KW, int Ktot,
                              int Ksteps, int Ntiles) {
    int id = blockIdx.x * blockDim.x + threadIdx.x;
    int total = Ntiles * Ksteps * 32;
    if (id >= total) return;
    int lane = id & 31;
    int tk = id >> 5;
    int ks = tk % Ksteps;
    int tn = tk / Ksteps;
    int n = tn * 16 + (lane & 15);
    __bf16 tmp[16];
#pragma unroll
    for (int e = 0; e < 16; ++e) {
        int k = ks * 32 + kmapB(lane, e);
        float v = 0.0f;
        if (n < Nout && k < Ktot) {
            int ci = k % Cin;
            int pos = k / Cin;
            int ky = pos / KW;
            int kx = pos % KW;
            v = src[((size_t)(n * Cin + ci) * KH + ky) * KW + kx];
        }
        tmp[e] = (__bf16)v;
    }
    *reinterpret_cast<v16bf*>(dst + (size_t)id * 16) = *reinterpret_cast<v16bf*>(tmp);
}

// ---------------------------------------------------------------------------
// Plain pack-A: src bf16 row-major [Mrows][Ktot] -> packed A fragments
// ---------------------------------------------------------------------------
__global__ void pack_a_plain(const __bf16* __restrict__ src, __bf16* __restrict__ dst,
                             int Mrows, int Ktot, int Ksteps, int Mtiles) {
    int id = blockIdx.x * blockDim.x + threadIdx.x;
    int total = Mtiles * Ksteps * 32;
    if (id >= total) return;
    int lane = id & 31;
    int tk = id >> 5;
    int ks = tk % Ksteps;
    int tm = tk / Ksteps;
    int m = tm * 16 + (lane & 15);
    __bf16 tmp[16];
#pragma unroll
    for (int e = 0; e < 16; ++e) {
        int k = ks * 32 + kmapA(lane, e);
        __bf16 v = (__bf16)0.0f;
        if (m < Mrows && k < Ktot) v = src[(size_t)m * Ktot + k];
        tmp[e] = v;
    }
    *reinterpret_cast<v16bf*>(dst + (size_t)id * 16) = *reinterpret_cast<v16bf*>(tmp);
}

// ---------------------------------------------------------------------------
// Conv1 im2col+pack from raw fp32 frames: x [NF][2][84][84], 8x8 stride 4 -> 20x20
// K = 64 (ky*8+kx). Row m = nloc*400 + oy*20 + ox, frame = f0 + nloc.
// ---------------------------------------------------------------------------
__global__ void pack_a_conv1(const float* __restrict__ x, __bf16* __restrict__ dst,
                             int f0, int chan, int Mtiles) {
    int id = blockIdx.x * blockDim.x + threadIdx.x;
    int total = Mtiles * 2 * 32;  // Ksteps = 2
    if (id >= total) return;
    int lane = id & 31;
    int tk = id >> 5;
    int ks = tk & 1;
    int tm = tk >> 1;
    int m = tm * 16 + (lane & 15);
    int nloc = m / 400;
    int p = m % 400;
    int oy = p / 20, ox = p % 20;
    const float* base = x + ((size_t)((f0 + nloc) * 2 + chan)) * 84 * 84;
    __bf16 tmp[16];
#pragma unroll
    for (int e = 0; e < 16; ++e) {
        int k = ks * 32 + kmapA(lane, e);   // k in [0,64)
        int ky = k >> 3, kx = k & 7;
        tmp[e] = (__bf16)base[(size_t)(oy * 4 + ky) * 84 + (ox * 4 + kx)];
    }
    *reinterpret_cast<v16bf*>(dst + (size_t)id * 16) = *reinterpret_cast<v16bf*>(tmp);
}

// ---------------------------------------------------------------------------
// Generic conv im2col+pack from bf16 NHWC activations (chunk-local).
// k-order: k = (ky*KW + kx)*Cin + ci
// ---------------------------------------------------------------------------
__global__ void pack_a_conv(const __bf16* __restrict__ src, __bf16* __restrict__ dst,
                            int Mtiles, int Ksteps, int Ktot,
                            int Hin, int Win, int Cin, int KW, int stride,
                            int OH, int OW) {
    int id = blockIdx.x * blockDim.x + threadIdx.x;
    int total = Mtiles * Ksteps * 32;
    if (id >= total) return;
    int lane = id & 31;
    int tk = id >> 5;
    int ks = tk % Ksteps;
    int tm = tk / Ksteps;
    int m = tm * 16 + (lane & 15);
    int hw = OH * OW;
    int nloc = m / hw;
    int p = m % hw;
    int oy = p / OW, ox = p % OW;
    const __bf16* base = src + (size_t)nloc * Hin * Win * Cin;
    __bf16 tmp[16];
#pragma unroll
    for (int e = 0; e < 16; ++e) {
        int k = ks * 32 + kmapA(lane, e);
        __bf16 v = (__bf16)0.0f;
        if (k < Ktot) {
            int ci = k % Cin;
            int pos = k / Cin;
            int ky = pos / KW, kx = pos % KW;
            v = base[((size_t)(oy * stride + ky) * Win + (ox * stride + kx)) * Cin + ci];
        }
        tmp[e] = v;
    }
    *reinterpret_cast<v16bf*>(dst + (size_t)id * 16) = *reinterpret_cast<v16bf*>(tmp);
}

// ---------------------------------------------------------------------------
// WMMA GEMM: D[M,N] = A[M,K] x B[K,N] (+bias). Each wave computes 32x32.
// mode 0: f32 out[m*ldout+n] = acc + bias[n]                   (no relu)
// mode 1: bf16 out[m*ldout+n] = relu(acc + bias[n])            (row-major / NHWC)
// mode 2: bf16 out[nloc*ldout + n*HW + p] = relu(...), m=nloc*HW+p  (NCHW flatten)
// ---------------------------------------------------------------------------
__device__ __forceinline__ void epilogue_tile(const v8f& acc, int tm, int tn, int lane,
                                              const float* __restrict__ bias, void* out,
                                              int mode, int HW, int ldout) {
    int nl = lane & 15;
    int mh = (lane >> 4) << 3;
#pragma unroll
    for (int r = 0; r < 8; ++r) {
        int m = tm * 16 + mh + r;
        int n = tn * 16 + nl;
        float v = acc[r] + bias[n];
        if (mode == 0) {
            ((float*)out)[(size_t)m * ldout + n] = v;
        } else {
            v = v > 0.0f ? v : 0.0f;
            if (mode == 1) {
                ((__bf16*)out)[(size_t)m * ldout + n] = (__bf16)v;
            } else {
                int nloc = m / HW, pp = m % HW;
                ((__bf16*)out)[(size_t)nloc * ldout + n * HW + pp] = (__bf16)v;
            }
        }
    }
}

__global__ void __launch_bounds__(256)
gemm_wmma(const __bf16* __restrict__ A, const __bf16* __restrict__ Bm,
          const float* __restrict__ bias, void* __restrict__ out,
          int Mtiles, int Ntiles, int Ksteps, int mode, int HW, int ldout) {
    int wid = (blockIdx.x * blockDim.x + threadIdx.x) >> 5;
    int lane = threadIdx.x & 31;
    int nRegN = Ntiles >> 1;
    int nRegM = Mtiles >> 1;
    if (wid >= nRegN * nRegM) return;
    int rn = wid % nRegN;
    int rm = wid / nRegN;
    int tm0 = rm * 2, tm1 = tm0 + 1;
    int tn0 = rn * 2, tn1 = tn0 + 1;

    v8f acc00 = {0.f, 0.f, 0.f, 0.f, 0.f, 0.f, 0.f, 0.f};
    v8f acc01 = acc00, acc10 = acc00, acc11 = acc00;

    for (int ks = 0; ks < Ksteps; ++ks) {
        v16bf a0 = *reinterpret_cast<const v16bf*>(
            A + (((size_t)tm0 * Ksteps + ks) * 32 + lane) * 16);
        v16bf a1 = *reinterpret_cast<const v16bf*>(
            A + (((size_t)tm1 * Ksteps + ks) * 32 + lane) * 16);
        v16bf b0 = *reinterpret_cast<const v16bf*>(
            Bm + (((size_t)tn0 * Ksteps + ks) * 32 + lane) * 16);
        v16bf b1 = *reinterpret_cast<const v16bf*>(
            Bm + (((size_t)tn1 * Ksteps + ks) * 32 + lane) * 16);
        acc00 = wmma_bf16(a0, b0, acc00);
        acc01 = wmma_bf16(a0, b1, acc01);
        acc10 = wmma_bf16(a1, b0, acc10);
        acc11 = wmma_bf16(a1, b1, acc11);
    }
    epilogue_tile(acc00, tm0, tn0, lane, bias, out, mode, HW, ldout);
    epilogue_tile(acc01, tm0, tn1, lane, bias, out, mode, HW, ldout);
    epilogue_tile(acc10, tm1, tn0, lane, bias, out, mode, HW, ldout);
    epilogue_tile(acc11, tm1, tn1, lane, bias, out, mode, HW, ldout);
}

// ---------------------------------------------------------------------------
// bsum = bih + bhh
// ---------------------------------------------------------------------------
__global__ void vec_add_kernel(const float* a, const float* b, float* o, int n) {
    int i = blockIdx.x * blockDim.x + threadIdx.x;
    if (i < n) o[i] = a[i] + b[i];
}

// ---------------------------------------------------------------------------
// LSTM scan with done-resets. One block (8 waves, 256 thr) per modality.
// gates_x (= x@Wih^T + bih + bhh) precomputed in gxs [mod][T*B][512] f32.
// Wave w owns hidden columns [16w,16w+16) across all 4 gates (i,f,g,o),
// so gate activations stay in registers; h lives in LDS as bf16 [32][128].
// ---------------------------------------------------------------------------
__global__ void __launch_bounds__(256)
lstm_scan(const float* __restrict__ gxs, const __bf16* __restrict__ whhP, int whhStride,
          const float* __restrict__ v_h0, const float* __restrict__ v_c0,
          const float* __restrict__ a_h0, const float* __restrict__ a_c0,
          const int* __restrict__ done, float* __restrict__ hsb) {
    int mod = blockIdx.x;
    const float* gx = gxs + (size_t)mod * NF_ * 512;
    const __bf16* Bw = whhP + (size_t)mod * whhStride;
    const float* h0 = mod ? a_h0 : v_h0;
    const float* c0 = mod ? a_c0 : v_c0;
    float* hsOut = hsb + (size_t)mod * NF_ * 128;

    __shared__ __bf16 hsh[32 * 128];

    int tid = threadIdx.x;
    int lane = tid & 31;
    int wave = tid >> 5;
    int nl = lane & 15;
    int hb = lane >> 4;
    int ncol = wave * 16 + nl;

    // cell state in registers: creg[mt][r] holds c[m][ncol], m = mt*16 + hb*8 + r
    float creg[2][8];
#pragma unroll
    for (int mt = 0; mt < 2; ++mt)
#pragma unroll
        for (int r = 0; r < 8; ++r) {
            int m = mt * 16 + hb * 8 + r;
            creg[mt][r] = c0[(size_t)m * 128 + ncol];
        }
    for (int i = tid; i < 4096; i += 256) hsh[i] = (__bf16)h0[i];
    __syncthreads();

    for (int t = 0; t < T_; ++t) {
        // ---- reset h,c where done[t][b] == 1 (before gemm, as in reference) ----
        {
            int base = tid * 16;        // 16 contiguous LDS elems, all in row m
            int m = base >> 7;
            if (done[t * 32 + m]) {
#pragma unroll
                for (int ii = 0; ii < 16; ++ii) hsh[base + ii] = (__bf16)0.0f;
            }
        }
#pragma unroll
        for (int mt = 0; mt < 2; ++mt)
#pragma unroll
            for (int r = 0; r < 8; ++r) {
                int m = mt * 16 + hb * 8 + r;
                if (done[t * 32 + m]) creg[mt][r] = 0.0f;
            }
        __syncthreads();

        // ---- h @ Whh^T via WMMA: M=32 (2 tiles), K=128 (4 ksteps), wave cols ----
        v8f zero = {0.f, 0.f, 0.f, 0.f, 0.f, 0.f, 0.f, 0.f};
        v8f acc[4][2];
#pragma unroll
        for (int g = 0; g < 4; ++g) { acc[g][0] = zero; acc[g][1] = zero; }

#pragma unroll
        for (int ks = 0; ks < 4; ++ks) {
            v16bf afr[2];
#pragma unroll
            for (int mt = 0; mt < 2; ++mt) {
                int m = mt * 16 + nl;
                int kb = ks * 32 + hb * 8;
                v16bf a;
#pragma unroll
                for (int e = 0; e < 8; ++e) {
                    a[e]     = hsh[m * 128 + kb + e];
                    a[e + 8] = hsh[m * 128 + kb + 16 + e];
                }
                afr[mt] = a;
            }
#pragma unroll
            for (int g = 0; g < 4; ++g) {
                int tn = g * 8 + wave;  // N-tile: gate g, column block = wave
                v16bf b = *reinterpret_cast<const v16bf*>(
                    Bw + (((size_t)tn * 4 + ks) * 32 + lane) * 16);
                acc[g][0] = wmma_bf16(afr[0], b, acc[g][0]);
                acc[g][1] = wmma_bf16(afr[1], b, acc[g][1]);
            }
        }
        __syncthreads();  // all LDS h reads done before h is rewritten

        // ---- elementwise cell update (torch gate order i,f,g,o) ----
#pragma unroll
        for (int mt = 0; mt < 2; ++mt)
#pragma unroll
            for (int r = 0; r < 8; ++r) {
                int m = mt * 16 + hb * 8 + r;
                size_t row = (size_t)(t * 32 + m) * 512;
                float gi = acc[0][mt][r] + gx[row + 0   + ncol];
                float gf = acc[1][mt][r] + gx[row + 128 + ncol];
                float gg = acc[2][mt][r] + gx[row + 256 + ncol];
                float go = acc[3][mt][r] + gx[row + 384 + ncol];
                float c = sigm(gf) * creg[mt][r] + sigm(gi) * tanhf(gg);
                creg[mt][r] = c;
                float h = sigm(go) * tanhf(c);
                hsOut[(size_t)(t * 32 + m) * 128 + ncol] = h;
                hsh[m * 128 + ncol] = (__bf16)h;
            }
        __syncthreads();  // h visible to everyone for next step
    }
}

// ---------------------------------------------------------------------------
// LayerNorm both modalities, sum, actor/critic heads. One block per row.
// ---------------------------------------------------------------------------
__global__ void __launch_bounds__(128)
heads_kernel(const float* __restrict__ hsb,
             const float* __restrict__ vg, const float* __restrict__ vb,
             const float* __restrict__ ag, const float* __restrict__ ab,
             const float* __restrict__ actor_w, const float* __restrict__ actor_b,
             const float* __restrict__ critic_w, const float* __restrict__ critic_b,
             float* __restrict__ out) {
    int r = blockIdx.x;
    int tid = threadIdx.x;
    __shared__ float sred[128];
    __shared__ float hid[128];

    float v = hsb[(size_t)r * 128 + tid];
    float a = hsb[(size_t)NF_ * 128 + (size_t)r * 128 + tid];

    auto block_sum = [&](float x) -> float {
        sred[tid] = x;
        __syncthreads();
        for (int s = 64; s > 0; s >>= 1) {
            if (tid < s) sred[tid] += sred[tid + s];
            __syncthreads();
        }
        float res = sred[0];
        __syncthreads();
        return res;
    };

    float vmu  = block_sum(v) * (1.0f / 128.0f);
    float vvar = block_sum(v * v) * (1.0f / 128.0f) - vmu * vmu;
    float amu  = block_sum(a) * (1.0f / 128.0f);
    float avar = block_sum(a * a) * (1.0f / 128.0f) - amu * amu;

    float vn = (v - vmu) * rsqrtf(vvar + 1e-5f) * vg[tid] + vb[tid];
    float an = (a - amu) * rsqrtf(avar + 1e-5f) * ag[tid] + ab[tid];
    hid[tid] = vn + an;
    __syncthreads();

    if (tid < 19) {
        float s;
        if (tid < 18) {
            s = actor_b[tid];
            for (int k = 0; k < 128; ++k) s += hid[k] * actor_w[tid * 128 + k];
        } else {
            s = critic_b[0];
            for (int k = 0; k < 128; ++k) s += hid[k] * critic_w[k];
        }
        out[(size_t)r * 19 + tid] = s;
    }
}

// ---------------------------------------------------------------------------
// Host orchestration
// ---------------------------------------------------------------------------
struct ModParams {
    const float *bhh, *bih, *whh, *wih;
    const float *c1b, *c1w, *c2b, *c2w, *c3b, *c3w, *fcb, *fcw;
    const float *lnb, *lng;
};

extern "C" void kernel_launch(void* const* d_in, const int* in_sizes, int n_in,
                              void* d_out, int out_size, void* d_ws, size_t ws_size,
                              hipStream_t stream) {
    (void)in_sizes; (void)n_in; (void)out_size; (void)ws_size;

    // Input order: setup_inputs() dict order, params flattened as a sorted pytree.
    const float* x    = (const float*)d_in[0];
    const int*   done = (const int*)d_in[1];
    const float* v_h0 = (const float*)d_in[2];
    const float* v_c0 = (const float*)d_in[3];
    const float* a_h0 = (const float*)d_in[4];
    const float* a_c0 = (const float*)d_in[5];
    const float* actor_b = (const float*)d_in[6];
    const float* actor_w = (const float*)d_in[7];
    ModParams am { (const float*)d_in[8],  (const float*)d_in[9],
                   (const float*)d_in[10], (const float*)d_in[11],
                   (const float*)d_in[12], (const float*)d_in[13],
                   (const float*)d_in[14], (const float*)d_in[15],
                   (const float*)d_in[16], (const float*)d_in[17],
                   (const float*)d_in[18], (const float*)d_in[19],
                   (const float*)d_in[20], (const float*)d_in[21] };
    const float* critic_b = (const float*)d_in[22];
    const float* critic_w = (const float*)d_in[23];
    ModParams vm { (const float*)d_in[24], (const float*)d_in[25],
                   (const float*)d_in[26], (const float*)d_in[27],
                   (const float*)d_in[28], (const float*)d_in[29],
                   (const float*)d_in[30], (const float*)d_in[31],
                   (const float*)d_in[32], (const float*)d_in[33],
                   (const float*)d_in[34], (const float*)d_in[35],
                   (const float*)d_in[36], (const float*)d_in[37] };
    ModParams mods[2] = { vm, am };  // mod 0 = video (channel 0), mod 1 = audio
    float* out = (float*)d_out;

    // ---- workspace carve-up (256B aligned) ----
    char* wp = (char*)d_ws;
    auto alloc = [&](size_t bytes) -> char* {
        char* r = wp;
        wp += (bytes + 255) & ~(size_t)255;
        return r;
    };
    // GEMM tile bookkeeping
    const int M1t = CH_ * 400 / 16, K1s = 2,  N1t = 2;    // conv1
    const int M2t = CH_ * 81 / 16,  K2s = 16, N2t = 4;    // conv2
    const int M3t = CH_ * 49 / 16,  K3s = 18, N3t = 4;    // conv3
    const int Mft = CH_ / 16,       Kfs = 98, Nft = 32;   // fc (K=3136)
    const int Kis = 16,             Nit = 32;             // wih (K=512)
    const int Khs = 4,              Nht = 32;             // whh (K=128)
    const size_t PB1 = (size_t)N1t * K1s * 512, PB2 = (size_t)N2t * K2s * 512;
    const size_t PB3 = (size_t)N3t * K3s * 512, PBf = (size_t)Nft * Kfs * 512;
    const size_t PBi = (size_t)Nit * Kis * 512, PBh = (size_t)Nht * Khs * 512;

    float*  gxs   = (float*)alloc((size_t)2 * NF_ * 512 * 4);
    float*  hsb   = (float*)alloc((size_t)2 * NF_ * 128 * 4);
    float*  bsum  = (float*)alloc((size_t)2 * 512 * 4);
    __bf16* WB1   = (__bf16*)alloc(2 * PB1 * 2);
    __bf16* WB2   = (__bf16*)alloc(2 * PB2 * 2);
    __bf16* WB3   = (__bf16*)alloc(2 * PB3 * 2);
    __bf16* WBf   = (__bf16*)alloc(2 * PBf * 2);
    __bf16* WBi   = (__bf16*)alloc(2 * PBi * 2);
    __bf16* WBh   = (__bf16*)alloc(2 * PBh * 2);
    __bf16* colA1 = (__bf16*)alloc((size_t)M1t * K1s * 512 * 2);
    __bf16* act1  = (__bf16*)alloc((size_t)CH_ * 400 * 32 * 2);
    __bf16* colA2 = (__bf16*)alloc((size_t)M2t * K2s * 512 * 2);
    __bf16* act2  = (__bf16*)alloc((size_t)CH_ * 81 * 64 * 2);
    __bf16* colA3 = (__bf16*)alloc((size_t)M3t * K3s * 512 * 2);
    __bf16* act3  = (__bf16*)alloc((size_t)CH_ * 3136 * 2);
    __bf16* colAf = (__bf16*)alloc((size_t)Mft * Kfs * 512 * 2);
    __bf16* feats = (__bf16*)alloc((size_t)CH_ * 512 * 2);
    __bf16* colAt = (__bf16*)alloc((size_t)Mft * Kis * 512 * 2);

    auto gemm = [&](const __bf16* A, const __bf16* Bm, const float* bias, void* o,
                    int Mtiles, int Ntiles, int Ksteps, int mode, int HW, int ld) {
        int regions = (Mtiles >> 1) * (Ntiles >> 1);
        gemm_wmma<<<DIVUP(regions, 8), 256, 0, stream>>>(A, Bm, bias, o, Mtiles,
                                                         Ntiles, Ksteps, mode, HW, ld);
    };

    for (int mod = 0; mod < 2; ++mod) {
        const ModParams& P = mods[mod];
        __bf16 *wb1 = WB1 + mod * PB1, *wb2 = WB2 + mod * PB2, *wb3 = WB3 + mod * PB3;
        __bf16 *wbf = WBf + mod * PBf, *wbi = WBi + mod * PBi, *wbh = WBh + mod * PBh;

        // ---- weight prepack ----
        pack_b_kernel<<<DIVUP(N1t * K1s * 32, 256), 256, 0, stream>>>(
            P.c1w, wb1, 32, 1, 8, 8, 64, K1s, N1t);
        pack_b_kernel<<<DIVUP(N2t * K2s * 32, 256), 256, 0, stream>>>(
            P.c2w, wb2, 64, 32, 4, 4, 512, K2s, N2t);
        pack_b_kernel<<<DIVUP(N3t * K3s * 32, 256), 256, 0, stream>>>(
            P.c3w, wb3, 64, 64, 3, 3, 576, K3s, N3t);
        pack_b_kernel<<<DIVUP(Nft * Kfs * 32, 256), 256, 0, stream>>>(
            P.fcw, wbf, 512, 3136, 1, 1, 3136, Kfs, Nft);
        pack_b_kernel<<<DIVUP(Nit * Kis * 32, 256), 256, 0, stream>>>(
            P.wih, wbi, 512, 512, 1, 1, 512, Kis, Nit);
        pack_b_kernel<<<DIVUP(Nht * Khs * 32, 256), 256, 0, stream>>>(
            P.whh, wbh, 512, 128, 1, 1, 128, Khs, Nht);
        vec_add_kernel<<<2, 256, 0, stream>>>(P.bih, P.bhh, bsum + mod * 512, 512);

        // ---- conv/FC/gates pipeline, chunked over frames ----
        for (int c = 0; c < NCHUNK_; ++c) {
            int f0 = c * CH_;
            pack_a_conv1<<<DIVUP(M1t * K1s * 32, 256), 256, 0, stream>>>(
                x, colA1, f0, mod, M1t);
            gemm(colA1, wb1, P.c1b, act1, M1t, N1t, K1s, 1, 400, 32);

            pack_a_conv<<<DIVUP(M2t * K2s * 32, 256), 256, 0, stream>>>(
                act1, colA2, M2t, K2s, 512, 20, 20, 32, 4, 2, 9, 9);
            gemm(colA2, wb2, P.c2b, act2, M2t, N2t, K2s, 1, 81, 64);

            pack_a_conv<<<DIVUP(M3t * K3s * 32, 256), 256, 0, stream>>>(
                act2, colA3, M3t, K3s, 576, 9, 9, 64, 3, 1, 7, 7);
            gemm(colA3, wb3, P.c3b, act3, M3t, N3t, K3s, 2, 49, 3136);

            pack_a_plain<<<DIVUP(Mft * Kfs * 32, 256), 256, 0, stream>>>(
                act3, colAf, CH_, 3136, Kfs, Mft);
            gemm(colAf, wbf, P.fcb, feats, Mft, Nft, Kfs, 1, 1, 512);

            pack_a_plain<<<DIVUP(Mft * Kis * 32, 256), 256, 0, stream>>>(
                feats, colAt, CH_, 512, Kis, Mft);
            float* gxChunk = gxs + (size_t)mod * NF_ * 512 + (size_t)f0 * 512;
            gemm(colAt, wbi, bsum + mod * 512, gxChunk, Mft, Nit, Kis, 0, 1, 512);
        }
    }

    // ---- recurrent scan (both modalities in parallel) ----
    lstm_scan<<<2, 256, 0, stream>>>(gxs, WBh, (int)PBh, v_h0, v_c0, a_h0, a_c0,
                                     done, hsb);

    // ---- layernorm + sum + heads ----
    heads_kernel<<<NF_, 128, 0, stream>>>(hsb, mods[0].lng, mods[0].lnb,
                                          mods[1].lng, mods[1].lnb,
                                          actor_w, actor_b, critic_w, critic_b, out);
}
